// Cond_NCP_CfC_9766755631842
// MI455X (gfx1250) — matrix operand order repaired and checked by reading, not compiled, not run on hardware
//
#include <hip/hip_runtime.h>
#include <cstddef>

// ---------- types ----------
typedef __attribute__((ext_vector_type(16))) __bf16   v16bf;
typedef __attribute__((ext_vector_type(8)))  float    v8f;
typedef __attribute__((ext_vector_type(4)))  unsigned u32x4;
typedef __attribute__((ext_vector_type(4)))  float    f32x4;
typedef __attribute__((ext_vector_type(4)))  int      v4i;

union Frag { u32x4 q[2]; v16bf v; };

// ---------- optional CDNA5 async-to-LDS path (compile-safe probes) ----------
#if defined(__has_builtin)
#  if __has_builtin(__builtin_amdgcn_global_load_async_to_lds_b128)
#    define HAVE_ASYNC_LDS 1
#  endif
#  if __has_builtin(__builtin_amdgcn_s_wait_asynccnt)
#    define HAVE_WAIT_ASYNC 1
#  endif
#endif

typedef __attribute__((address_space(1))) v4i* as1_v4i;
typedef __attribute__((address_space(3))) v4i* as3_v4i;

// ---------- problem constants ----------
#define NWG   10
#define BSZ   128
#define TT    512
#define NFEAT 508
#define NCMD  4

#define D0 152
#define D1 101
#define D2 3
#define K0 672     // 512 (x) + 160 (h0 pad)
#define K1 288     // 160 (h0 pad) + 128 (h1 pad)
#define K2 160     // 128 (h1 pad) + 32 (h2 pad)
#define KC0 21
#define KC1 9
#define KC2 5
#define H0ROW 160
#define H1ROW 128
#define H2ROW 32
#define NT1 7

// ---------- workspace layout (bytes) ----------
#define OFF_H0  256
#define OFF_H1  (OFF_H0 + 2*BSZ*H0ROW*2)            // 82176
#define OFF_H2  (OFF_H1 + 2*BSZ*H1ROW*2)            // 147712
#define OFF_W0  (OFF_H2 + 2*BSZ*H2ROW*2)            // 164096
#define OFF_W1  (OFF_W0 + NWG*3*K0*16*2)            // 809216
#define OFF_W2  (OFF_W1 + NT1*3*K1*16*2)            // 1002752
#define WS_END  (OFF_W2 + 3*K2*16*2)                // 1018112
#define OFF_XBF WS_END                               // 256-aligned
#define XBF_BYTES ((size_t)BSZ * TT * 512 * 2)       // 67 MB packed bf16 inputs
#define WS_BIG  ((size_t)OFF_XBF + XBF_BYTES)

#define LDSB    (3*K0*16*2 + 3*K1*16*2)             // 92160 B dynamic LDS

// ---------- small helpers ----------
static __device__ __forceinline__ unsigned short f2bf(float f) {
  unsigned u = __float_as_uint(f);
  unsigned r = u + 0x7FFFu + ((u >> 16) & 1u);      // RNE
  return (unsigned short)(r >> 16);
}
static __device__ __forceinline__ unsigned pk_trunc(float a, float b) {
  return (__float_as_uint(a) >> 16) | (__float_as_uint(b) & 0xFFFF0000u);
}
static __device__ __forceinline__ float fast_tanh(float x) {
  x = fminf(fmaxf(x, -15.f), 15.f);
  float e = __expf(2.0f * x);
  return (e - 1.0f) / (e + 1.0f);
}
static __device__ __forceinline__ float fast_sig(float x) {
  return 1.0f / (1.0f + __expf(-x));
}
static __device__ __forceinline__ u32x4 ld_bf8(const unsigned short* p) {
  return *reinterpret_cast<const u32x4*>(p);
}
// B fragment: lane l holds column n = l%16, K in table order (8h+s, 16+8h+s-8), gathered from [k][16] LDS
static __device__ __forceinline__ Frag loadB(const unsigned short* ldsw, int kc, int lane) {
  const int n = lane & 15, kb = kc * 32 + (lane >> 4) * 8;
  Frag f;
#pragma unroll
  for (int j = 0; j < 8; ++j) {
    const int k = kb + (j & 3) * 2 + (j >> 2) * 16;
    f.q[j >> 2][j & 3] = (unsigned)ldsw[(size_t)k * 16 + n] | ((unsigned)ldsw[(size_t)(k + 1) * 16 + n] << 16);
  }
  return f;
}
static __device__ __forceinline__ v8f wmma_bf(v16bf a, v16bf b, v8f c) {
  return __builtin_amdgcn_wmma_f32_16x16x32_bf16(false, a, false, b, (short)0, c, false, false);
}

// ---------- async (or fallback) staging of weights into LDS ----------
static __device__ __forceinline__ void stage_to_lds(unsigned short* ldsdst,
                                                    const unsigned short* src,
                                                    int n_u16, int tid) {
  const int n16 = n_u16 / 8;  // 16B chunks
#if defined(HAVE_ASYNC_LDS)
  for (int i = tid; i < n16; i += 256) {
    as1_v4i g = (as1_v4i)(size_t)(src + (size_t)i * 8);
    as3_v4i l = (as3_v4i)(unsigned)(size_t)(ldsdst + (size_t)i * 8);
    __builtin_amdgcn_global_load_async_to_lds_b128(g, l, 0, 0);
  }
#else
  u32x4* dst = reinterpret_cast<u32x4*>(ldsdst);
  const u32x4* s = reinterpret_cast<const u32x4*>(src);
  for (int i = tid; i < n16; i += 256) dst[i] = s[i];
#endif
}
static __device__ __forceinline__ void stage_wait() {
#if defined(HAVE_ASYNC_LDS)
#  if defined(HAVE_WAIT_ASYNC)
  __builtin_amdgcn_s_wait_asynccnt(0);
#  else
  asm volatile("s_wait_asynccnt 0" ::: "memory");
#  endif
#endif
}

// ---------- grid barrier (10 co-resident WGs) ----------
static __device__ __forceinline__ void grid_barrier(unsigned* cnt, unsigned* gen, unsigned target) {
  __threadfence();
  __syncthreads();
  if (threadIdx.x == 0) {
    if (atomicAdd(cnt, 1u) == NWG - 1) {
      atomicExch(cnt, 0u);
      __threadfence();
      atomicAdd(gen, 1u);
    } else {
      while (atomicAdd(gen, 0u) < target) __builtin_amdgcn_s_sleep(2);
    }
  }
  __syncthreads();
  __threadfence();
}

// ---------- A-fragment group gathers (8 bf16 = 16B per call) ----------
template<bool PK>
static __device__ __forceinline__ u32x4 a0grp(const float* feat, const float* cmd,
                                              const unsigned short* xrow,
                                              const unsigned short* h0r, int b, int t, int kg) {
  if (kg >= 512) return ld_bf8(h0r + (size_t)b * H0ROW + (kg - 512));
  if (PK)        return ld_bf8(xrow + kg);
  f32x4 lo, hi;
  const float* p = feat + ((size_t)b * TT + t) * NFEAT;
  if (kg < 504) {
    lo = *reinterpret_cast<const f32x4*>(p + kg);
    hi = *reinterpret_cast<const f32x4*>(p + kg + 4);
  } else {
    lo = *reinterpret_cast<const f32x4*>(p + 504);
    hi = *reinterpret_cast<const f32x4*>(cmd + ((size_t)b * TT + t) * NCMD);
  }
  u32x4 r;
  r.x = pk_trunc(lo.x, lo.y); r.y = pk_trunc(lo.z, lo.w);
  r.z = pk_trunc(hi.x, hi.y); r.w = pk_trunc(hi.z, hi.w);
  return r;
}
static __device__ __forceinline__ u32x4 a1grp(const unsigned short* h0w, const unsigned short* h1r,
                                              int b, int kg) {
  if (kg < 160) return ld_bf8(h0w + (size_t)b * H0ROW + kg);
  return ld_bf8(h1r + (size_t)b * H1ROW + (kg - 160));
}
static __device__ __forceinline__ u32x4 a2grp(const unsigned short* h1w, const unsigned short* h2r,
                                              int b, int kg) {
  if (kg < 128) return ld_bf8(h1w + (size_t)b * H1ROW + kg);
  return ld_bf8(h2r + (size_t)b * H2ROW + (kg - 128));
}

// ---------- per-layer compute ----------
template<bool PK>
static __device__ __forceinline__ void run_layer0(
    const float* feat, const float* cmd, const unsigned short* xbf, const float* bias,
    const unsigned short* h0r, unsigned short* h0w, float* hx,
    const unsigned short* ldsw, int ntile, int lane, int wave, int t)
{
  v8f a0 = {0,0,0,0,0,0,0,0}, a1 = a0, a2 = a0;
  const int brow0 = wave * 16;
  const int b = brow0 + (lane & 15);
  const int ksel = (lane >> 4) * 8;
  const unsigned short* xrow = PK ? (xbf + ((size_t)b * TT + t) * 512) : (const unsigned short*)0;
  if (t + 1 < TT) {
    if (PK) __builtin_prefetch(xrow + 512, 0, 1);
    else    __builtin_prefetch(feat + ((size_t)b * TT + (t + 1)) * NFEAT, 0, 1);
  }
  Frag A;
  A.q[0] = a0grp<PK>(feat, cmd, xrow, h0r, b, t, ksel);
  A.q[1] = a0grp<PK>(feat, cmd, xrow, h0r, b, t, ksel + 16);
  for (int kc = 0; kc < KC0; ++kc) {
    const Frag B0 = loadB(ldsw,               kc, lane);
    const Frag B1 = loadB(ldsw + K0 * 16,     kc, lane);
    const Frag B2 = loadB(ldsw + 2 * K0 * 16, kc, lane);
    Frag An = A;
    if (kc + 1 < KC0) {
      const int kA = (kc + 1) * 32 + ksel;
      An.q[0] = a0grp<PK>(feat, cmd, xrow, h0r, b, t, kA);
      An.q[1] = a0grp<PK>(feat, cmd, xrow, h0r, b, t, kA + 16);
    }
    a0 = wmma_bf(A.v, B0.v, a0);
    a1 = wmma_bf(A.v, B1.v, a1);
    a2 = wmma_bf(A.v, B2.v, a2);
    A = An;
  }
  const int n = ntile * 16 + (lane & 15);
  const int mofs = (lane >> 4) * 8;
  const bool valid = (n < D0);
  const float b0v = valid ? bias[n] : 0.f;
  const float b1v = valid ? bias[D0 + n] : 0.f;
  const float b2v = valid ? (bias[2 * D0 + n] + bias[3 * D0 + n]) : 0.f;
#pragma unroll
  for (int r = 0; r < 8; ++r) {
    const float s = fast_sig(a2[r] + b2v);
    const float h = fast_tanh(a0[r] + b0v) * (1.f - s) + s * fast_tanh(a1[r] + b1v);
    if (valid) {
      const int br = brow0 + mofs + r;
      h0w[(size_t)br * H0ROW + n] = f2bf(h);
      if (t == TT - 1) hx[(size_t)br * 256 + n] = h;
    }
  }
}

static __device__ __forceinline__ void run_layer1(
    const float* bias, const unsigned short* h0w, const unsigned short* h1r,
    unsigned short* h1w, float* hx, const unsigned short* ldsw,
    int ntile, int lane, int wave, int t)
{
  v8f a0 = {0,0,0,0,0,0,0,0}, a1 = a0, a2 = a0;
  const int brow0 = wave * 16;
  const int b = brow0 + (lane & 15);
  const int ksel = (lane >> 4) * 8;
  Frag A;
  A.q[0] = a1grp(h0w, h1r, b, ksel);
  A.q[1] = a1grp(h0w, h1r, b, ksel + 16);
#pragma unroll
  for (int kc = 0; kc < KC1; ++kc) {
    const Frag B0 = loadB(ldsw,               kc, lane);
    const Frag B1 = loadB(ldsw + K1 * 16,     kc, lane);
    const Frag B2 = loadB(ldsw + 2 * K1 * 16, kc, lane);
    Frag An = A;
    if (kc + 1 < KC1) {
      const int kA = (kc + 1) * 32 + ksel;
      An.q[0] = a1grp(h0w, h1r, b, kA);
      An.q[1] = a1grp(h0w, h1r, b, kA + 16);
    }
    a0 = wmma_bf(A.v, B0.v, a0);
    a1 = wmma_bf(A.v, B1.v, a1);
    a2 = wmma_bf(A.v, B2.v, a2);
    A = An;
  }
  const int n = ntile * 16 + (lane & 15);
  const int mofs = (lane >> 4) * 8;
  const bool valid = (n < D1);
  const float b0v = valid ? bias[n] : 0.f;
  const float b1v = valid ? bias[D1 + n] : 0.f;
  const float b2v = valid ? (bias[2 * D1 + n] + bias[3 * D1 + n]) : 0.f;
#pragma unroll
  for (int r = 0; r < 8; ++r) {
    const float s = fast_sig(a2[r] + b2v);
    const float h = fast_tanh(a0[r] + b0v) * (1.f - s) + s * fast_tanh(a1[r] + b1v);
    if (valid) {
      const int br = brow0 + mofs + r;
      h1w[(size_t)br * H1ROW + n] = f2bf(h);
      if (t == TT - 1) hx[(size_t)br * 256 + D0 + n] = h;
    }
  }
}

static __device__ __forceinline__ void run_layer2(
    const float* bias, const unsigned short* h1w, const unsigned short* h2r,
    unsigned short* h2w, float* hx, float* out, const unsigned short* ldsw,
    int lane, int wave, int t)
{
  v8f a0 = {0,0,0,0,0,0,0,0}, a1 = a0, a2 = a0;
  const int brow0 = wave * 16;
  const int b = brow0 + (lane & 15);
  const int ksel = (lane >> 4) * 8;
#pragma unroll
  for (int kc = 0; kc < KC2; ++kc) {
    const int kA = kc * 32 + ksel;
    Frag A;
    A.q[0] = a2grp(h1w, h2r, b, kA);
    A.q[1] = a2grp(h1w, h2r, b, kA + 16);
    const Frag B0 = loadB(ldsw,               kc, lane);
    const Frag B1 = loadB(ldsw + K2 * 16,     kc, lane);
    const Frag B2 = loadB(ldsw + 2 * K2 * 16, kc, lane);
    a0 = wmma_bf(A.v, B0.v, a0);
    a1 = wmma_bf(A.v, B1.v, a1);
    a2 = wmma_bf(A.v, B2.v, a2);
  }
  const int n = lane & 15;
  const int mofs = (lane >> 4) * 8;
  const bool valid = (n < D2);
  const float b0v = valid ? bias[n] : 0.f;
  const float b1v = valid ? bias[D2 + n] : 0.f;
  const float b2v = valid ? (bias[2 * D2 + n] + bias[3 * D2 + n]) : 0.f;
#pragma unroll
  for (int r = 0; r < 8; ++r) {
    const float s = fast_sig(a2[r] + b2v);
    const float h = fast_tanh(a0[r] + b0v) * (1.f - s) + s * fast_tanh(a1[r] + b1v);
    if (valid) {
      const int br = brow0 + mofs + r;
      h2w[(size_t)br * H2ROW + n] = f2bf(h);
      const float c = (n == 0) ? fast_tanh(h) : fast_sig(h);
      out[((size_t)br * TT + t) * 3 + n] = c;
      if (t == TT - 1) hx[(size_t)br * 256 + 253 + n] = h;
    }
  }
}

// ---------- prep kernel: masks, Wa+Wb fuse, bf16 K-major weight tiles, zero H/barrier ----------
extern "C" __global__ void cfc_prep(
    const float* __restrict__ l0W, const float* __restrict__ l1W, const float* __restrict__ l2W,
    const float* __restrict__ m0, const float* __restrict__ m1, const float* __restrict__ m2,
    char* __restrict__ ws)
{
  unsigned* z = reinterpret_cast<unsigned*>(ws);
  unsigned short* w0 = reinterpret_cast<unsigned short*>(ws + OFF_W0);
  unsigned short* w1 = reinterpret_cast<unsigned short*>(ws + OFF_W1);
  unsigned short* w2 = reinterpret_cast<unsigned short*>(ws + OFF_W2);
  const int ZW = OFF_W0 / 4;
  const int E0 = NWG * 3 * K0 * 16;
  const int E1 = NT1 * 3 * K1 * 16;
  const int E2 = 3 * K2 * 16;
  const int total = ZW + E0 + E1 + E2;
  for (int i = blockIdx.x * blockDim.x + threadIdx.x; i < total; i += gridDim.x * blockDim.x) {
    if (i < ZW) { z[i] = 0u; continue; }
    int e = i - ZW;
    float val = 0.f;
    if (e < E0) {
      const int tile = e / (3 * K0 * 16);
      int r = e - tile * (3 * K0 * 16);
      const int mat = r / (K0 * 16); r -= mat * (K0 * 16);
      const int k = r >> 4, nn = r & 15;
      const int n = tile * 16 + nn;
      if (n < D0 && k < 664) {
        if (mat < 2) val = l0W[((size_t)mat * D0 + n) * 664 + k] * m0[(size_t)n * 664 + k];
        else         val = l0W[((size_t)2 * D0 + n) * 664 + k] + l0W[((size_t)3 * D0 + n) * 664 + k];
      }
      w0[e] = f2bf(val);
    } else if (e < E0 + E1) {
      e -= E0;
      const int tile = e / (3 * K1 * 16);
      int r = e - tile * (3 * K1 * 16);
      const int mat = r / (K1 * 16); r -= mat * (K1 * 16);
      const int k = r >> 4, nn = r & 15;
      const int n = tile * 16 + nn;
      int kin = -1;
      if (k < 160) { if (k < D0) kin = k; }
      else         { if (k - 160 < D1) kin = D0 + (k - 160); }
      if (n < D1 && kin >= 0) {
        if (mat < 2) val = l1W[((size_t)mat * D1 + n) * 253 + kin] * m1[(size_t)n * 253 + kin];
        else         val = l1W[((size_t)2 * D1 + n) * 253 + kin] + l1W[((size_t)3 * D1 + n) * 253 + kin];
      }
      w1[e] = f2bf(val);
    } else {
      e -= E0 + E1;
      const int mat = e / (K2 * 16);
      int r = e - mat * (K2 * 16);
      const int k = r >> 4, nn = r & 15;
      int kin = -1;
      if (k < 128) { if (k < D1) kin = k; }
      else         { if (k - 128 < D2) kin = D1 + (k - 128); }
      if (nn < D2 && kin >= 0) {
        if (mat < 2) val = l2W[((size_t)mat * D2 + nn) * 104 + kin] * m2[(size_t)nn * 104 + kin];
        else         val = l2W[((size_t)2 * D2 + nn) * 104 + kin] + l2W[((size_t)3 * D2 + nn) * 104 + kin];
      }
      w2[e] = f2bf(val);
    }
  }
}

// ---------- pack kernel: x inputs -> bf16 [B][T][512] (only when ws is large enough) ----------
extern "C" __global__ void cfc_pack(const float* __restrict__ feat, const float* __restrict__ cmd,
                                    unsigned short* __restrict__ xbf)
{
  const size_t total = (size_t)BSZ * TT * 512;
  for (size_t i = (size_t)blockIdx.x * blockDim.x + threadIdx.x; i < total;
       i += (size_t)gridDim.x * blockDim.x) {
    const size_t bt = i >> 9;
    const int k = (int)(i & 511);
    const float v = (k < NFEAT) ? feat[bt * NFEAT + k] : cmd[bt * NCMD + (k - NFEAT)];
    xbf[i] = f2bf(v);
  }
}

// ---------- persistent cooperative main kernel ----------
template<bool PK>
static __device__ __forceinline__ void cfc_main_dev(
    const float* feat, const float* cmd, const unsigned short* xbf,
    const float* b0, const float* b1, const float* b2,
    char* ws, float* out, unsigned short* lds)
{
  const int wg = blockIdx.x;
  const int tid = threadIdx.x;
  const int wave = tid >> 5, lane = tid & 31;

  unsigned* bar_cnt = reinterpret_cast<unsigned*>(ws);
  unsigned* bar_gen = bar_cnt + 1;
  unsigned short* H0 = reinterpret_cast<unsigned short*>(ws + OFF_H0);
  unsigned short* H1 = reinterpret_cast<unsigned short*>(ws + OFF_H1);
  unsigned short* H2 = reinterpret_cast<unsigned short*>(ws + OFF_H2);
  const unsigned short* W0g = reinterpret_cast<const unsigned short*>(ws + OFF_W0);
  const unsigned short* W1g = reinterpret_cast<const unsigned short*>(ws + OFF_W1);
  const unsigned short* W2g = reinterpret_cast<const unsigned short*>(ws + OFF_W2);

  // stage this WG's weight tiles into LDS (persist for all 512 steps)
  stage_to_lds(lds, W0g + (size_t)wg * 3 * K0 * 16, 3 * K0 * 16, tid);
  if (wg < NT1)           stage_to_lds(lds + 3 * K0 * 16, W1g + (size_t)wg * 3 * K1 * 16, 3 * K1 * 16, tid);
  else if (wg == NWG - 1) stage_to_lds(lds + 3 * K0 * 16, W2g, 3 * K2 * 16, tid);
  stage_wait();
  __syncthreads();
  const unsigned short* lds0 = lds;
  const unsigned short* ldsx = lds + 3 * K0 * 16;

  float* hx = out + (size_t)BSZ * TT * 3;
  unsigned phase = 0;

  for (int t = 0; t < TT; ++t) {
    const int w = t & 1, rp = w ^ 1;
    unsigned short*       h0w = H0 + (size_t)w  * BSZ * H0ROW;
    const unsigned short* h0r = H0 + (size_t)rp * BSZ * H0ROW;
    unsigned short*       h1w = H1 + (size_t)w  * BSZ * H1ROW;
    const unsigned short* h1r = H1 + (size_t)rp * BSZ * H1ROW;
    unsigned short*       h2w = H2 + (size_t)w  * BSZ * H2ROW;
    const unsigned short* h2r = H2 + (size_t)rp * BSZ * H2ROW;

    run_layer0<PK>(feat, cmd, xbf, b0, h0r, h0w, hx, lds0, wg, lane, wave, t);
    grid_barrier(bar_cnt, bar_gen, ++phase);
    if (wg < NT1) run_layer1(b1, h0w, h1r, h1w, hx, ldsx, wg, lane, wave, t);
    grid_barrier(bar_cnt, bar_gen, ++phase);
    if (wg == NWG - 1) run_layer2(b2, h1w, h2r, h2w, hx, out, ldsx, lane, wave, t);
  }
}

extern "C" __global__ __launch_bounds__(256, 1)
void cfc_main_pk(const float* __restrict__ feat, const float* __restrict__ cmd,
                 const unsigned short* __restrict__ xbf,
                 const float* __restrict__ b0, const float* __restrict__ b1,
                 const float* __restrict__ b2, char* __restrict__ ws,
                 float* __restrict__ out)
{
  extern __shared__ unsigned short lds[];
  cfc_main_dev<true>(feat, cmd, xbf, b0, b1, b2, ws, out, lds);
}

extern "C" __global__ __launch_bounds__(256, 1)
void cfc_main_cv(const float* __restrict__ feat, const float* __restrict__ cmd,
                 const unsigned short* __restrict__ xbf,
                 const float* __restrict__ b0, const float* __restrict__ b1,
                 const float* __restrict__ b2, char* __restrict__ ws,
                 float* __restrict__ out)
{
  extern __shared__ unsigned short lds[];
  cfc_main_dev<false>(feat, cmd, xbf, b0, b1, b2, ws, out, lds);
}

// ---------- host launch ----------
extern "C" void kernel_launch(void* const* d_in, const int* in_sizes, int n_in,
                              void* d_out, int out_size, void* d_ws, size_t ws_size,
                              hipStream_t stream) {
  (void)in_sizes; (void)n_in; (void)out_size;
  const float* feat = (const float*)d_in[0];
  const float* cmd  = (const float*)d_in[1];
  const float* l0W  = (const float*)d_in[2];
  const float* l0b  = (const float*)d_in[3];
  const float* l1W  = (const float*)d_in[4];
  const float* l1b  = (const float*)d_in[5];
  const float* l2W  = (const float*)d_in[6];
  const float* l2b  = (const float*)d_in[7];
  const float* m0   = (const float*)d_in[8];
  const float* m1   = (const float*)d_in[9];
  const float* m2   = (const float*)d_in[10];
  char* ws = (char*)d_ws;

  cfc_prep<<<512, 256, 0, stream>>>(l0W, l1W, l2W, m0, m1, m2, ws);

  const bool pk = (ws_size >= WS_BIG);
  unsigned short* xbf = (unsigned short*)(ws + OFF_XBF);
  if (pk) {
    cfc_pack<<<2048, 256, 0, stream>>>(feat, cmd, xbf);
    (void)hipFuncSetAttribute((const void*)cfc_main_pk,
                              hipFuncAttributeMaxDynamicSharedMemorySize, LDSB);
    cfc_main_pk<<<NWG, 256, LDSB, stream>>>(feat, cmd, xbf, l0b, l1b, l2b, ws, (float*)d_out);
  } else {
    (void)hipFuncSetAttribute((const void*)cfc_main_cv,
                              hipFuncAttributeMaxDynamicSharedMemorySize, LDSB);
    cfc_main_cv<<<NWG, 256, LDSB, stream>>>(feat, cmd, xbf, l0b, l1b, l2b, ws, (float*)d_out);
  }
}
